// MultiHeadedSelfAttentionModule_52398601011269
// MI455X (gfx1250) — matrix operand order repaired and checked by reading, not compiled, not run on hardware
//
#include <hip/hip_runtime.h>
#include <hip/hip_bf16.h>

// ---------------------------------------------------------------------------
// MI455X (gfx1250, wave32) Transformer-XL relative MHSA, bf16 WMMA everywhere.
// B=8 S=2048 D=256 H=4 dh=64.  All matmuls use v_wmma_f32_16x16x32_bf16.
// Attention is flash-style: [B,H,S,S] scores never hit memory; rel_shift is
// resolved analytically (m = S-1-i+j) with on-the-fly WMMA pos tiles + LDS pick.
// ---------------------------------------------------------------------------

typedef __bf16 bf16_t;
typedef __attribute__((ext_vector_type(16))) __bf16 v16bf;
typedef __attribute__((ext_vector_type(8)))  float  v8f;

#define SEQ   2048
#define DM    256
#define NH    4
#define DH    64
#define ROWS  16384   // B*S

__device__ __forceinline__ v8f wmma_bf16(v16bf a, v16bf b, v8f c) {
  return __builtin_amdgcn_wmma_f32_16x16x32_bf16(false, a, false, b, (short)0, c, false, false);
}

// 16 bf16 fragment from two 8-element (16B) contiguous chunks
__device__ __forceinline__ v16bf ldfrag2(const bf16_t* p0, const bf16_t* p1) {
  union { uint4 u; __bf16 h[8]; } a, b;
  a.u = *(const uint4*)p0;
  b.u = *(const uint4*)p1;
  v16bf r;
#pragma unroll
  for (int e = 0; e < 8; ++e) { r[e] = a.h[e]; r[e + 8] = b.h[e]; }
  return r;
}

// A-fragment (16xK tile, K-chunk kc of 32): lane = row, 16 contiguous K at k0 + 16*half
__device__ __forceinline__ v16bf load_afrag(const bf16_t* base, int row, int kc,
                                            int rowStride, int lane) {
  const bf16_t* p = base + (size_t)row * rowStride + 32 * kc + ((lane & 16) ? 16 : 0);
  return ldfrag2(p, p + 8);
}

// B-fragment: lane = output column; 16 contiguous K elements starting at rowptr
__device__ __forceinline__ v16bf load_bfrag(const bf16_t* rowptr) {
  return ldfrag2(rowptr, rowptr + 8);
}

// ------------------------------- LayerNorm ---------------------------------
__global__ __launch_bounds__(256) void ln_kernel(const float* __restrict__ x,
                                                 const float* __restrict__ gamma,
                                                 const float* __restrict__ beta,
                                                 bf16_t* __restrict__ xln) {
  int lane = threadIdx.x & 31;
  int row  = (blockIdx.x * blockDim.x + threadIdx.x) >> 5;
  if (row >= ROWS) return;
  const float* xr = x + (size_t)row * DM;
  float v[8]; float s = 0.f;
#pragma unroll
  for (int e = 0; e < 8; ++e) { v[e] = xr[e * 32 + lane]; s += v[e]; }
#pragma unroll
  for (int m = 1; m < 32; m <<= 1) s += __shfl_xor(s, m, 32);
  float mu = s * (1.f / DM);
  float vs = 0.f;
#pragma unroll
  for (int e = 0; e < 8; ++e) { float d = v[e] - mu; vs += d * d; }
#pragma unroll
  for (int m = 1; m < 32; m <<= 1) vs += __shfl_xor(vs, m, 32);
  float rstd = rsqrtf(vs * (1.f / DM) + 1e-5f);
#pragma unroll
  for (int e = 0; e < 8; ++e) {
    int c = e * 32 + lane;
    xln[(size_t)row * DM + c] = (bf16_t)((v[e] - mu) * rstd * gamma[c] + beta[c]);
  }
}

// --------------------------- Positional encoding ---------------------------
__global__ __launch_bounds__(256) void pe_kernel(bf16_t* __restrict__ pe) {
  int idx = blockIdx.x * blockDim.x + threadIdx.x;
  if (idx >= SEQ * DM) return;
  int s = idx >> 8, d = idx & 255;
  int i2 = d >> 1;
  float div = __expf((float)(2 * i2) * (-9.210340371976184f / (float)DM)); // ln(1e4)
  float ang = (float)s * div;
  pe[idx] = (bf16_t)((d & 1) ? cosf(ang) : sinf(ang));
}

// ----------------------- Weight transpose + bf16 cast ----------------------
__global__ __launch_bounds__(256) void wtrans_kernel(
    const float* __restrict__ W0, const float* __restrict__ W1,
    const float* __restrict__ W2, const float* __restrict__ W3,
    const float* __restrict__ W4,
    bf16_t* __restrict__ T0, bf16_t* __restrict__ T1, bf16_t* __restrict__ T2,
    bf16_t* __restrict__ T3, bf16_t* __restrict__ T4) {
  int idx = blockIdx.x * blockDim.x + threadIdx.x;
  if (idx >= DM * DM) return;
  int w = blockIdx.y;
  const float* W = (w == 0) ? W0 : (w == 1) ? W1 : (w == 2) ? W2 : (w == 3) ? W3 : W4;
  bf16_t* T = (w == 0) ? T0 : (w == 1) ? T1 : (w == 2) ? T2 : (w == 3) ? T3 : T4;
  int n = idx >> 8, k = idx & 255;
  T[(size_t)n * DM + k] = (bf16_t)W[(size_t)k * DM + n];  // WT[n,k] = W[k,n]
}

// --------------------- Projection GEMM (M x 256 x 256) ---------------------
// One wave per 16x16 tile.  WT is pre-transposed bf16 (row n = column n of W).
__global__ __launch_bounds__(128) void proj_gemm_kernel(
    const bf16_t* __restrict__ X, const bf16_t* __restrict__ WT,
    const float* __restrict__ bmain, const float* __restrict__ bex0,
    const float* __restrict__ bex1,
    bf16_t* __restrict__ out0, bf16_t* __restrict__ out1,
    float* __restrict__ foutp, int M, int transp) {
  int lane = threadIdx.x & 31;
  int wid  = (blockIdx.x * blockDim.x + threadIdx.x) >> 5;
  int total = (M >> 4) * 16;
  if (wid >= total) return;
  int mt = wid >> 4, nt = wid & 15;
  int row0 = mt << 4, n0 = nt << 4;
  int lm = lane & 15;
  int hi = (lane & 16) ? 1 : 0;

  int rA = row0 + lm;
  int nB = n0 + lm;
  v8f acc = {};
#pragma unroll
  for (int kc = 0; kc < 8; ++kc) {
    v16bf a = load_afrag(X, rA, kc, DM, lane);
    const bf16_t* bp = WT + (size_t)nB * DM + 32 * kc + (hi ? 16 : 0);
    acc = wmma_bf16(a, load_bfrag(bp), acc);
  }
  int half8 = hi ? 8 : 0;
  int col = n0 + lm;
  float bm = bmain ? bmain[col] : 0.f;
  float b0 = bex0 ? bex0[col] : 0.f;
  float b1 = bex1 ? bex1[col] : 0.f;
#pragma unroll
  for (int r = 0; r < 8; ++r) {
    int row = row0 + r + half8;
    float v = acc[r] + bm;
    if (foutp) foutp[(size_t)row * DM + col] = v;
    if (out0) {
      float v0 = v + b0;
      if (transp) {  // v stored as vT[B][H][dh][S] for contiguous attn.V B-frags
        int b = row >> 11, s = row & 2047;
        out0[(size_t)((b * NH + (col >> 6)) * DH + (col & 63)) * SEQ + s] = (bf16_t)v0;
      } else {
        out0[(size_t)row * DM + col] = (bf16_t)v0;
      }
    }
    if (out1) out1[(size_t)row * DM + col] = (bf16_t)(v + b1);
  }
}

// --------------------------- Fused flash attention -------------------------
// One wave per (b, h, 16-row i-tile). Online softmax over all 2048 columns.
__global__ __launch_bounds__(128) void attn_kernel(
    const bf16_t* __restrict__ qu, const bf16_t* __restrict__ qv,
    const bf16_t* __restrict__ kb, const bf16_t* __restrict__ vT,
    const bf16_t* __restrict__ pb, bf16_t* __restrict__ ctx) {
  __shared__ float  ldsT[4][2][16][33];  // [wave][low/up][row][cc]
  __shared__ __bf16 ldsP[4][16][32];     // attn probs staging (C-frag -> A-frag)

  int lane = threadIdx.x & 31;
  int wsl  = threadIdx.x >> 5;
  int wid  = blockIdx.x * 4 + wsl;       // 0..4095
  int b  = wid >> 9;
  int h  = (wid >> 7) & 3;
  int i0 = (wid & 127) << 4;

  const bf16_t* qu_bh = qu + (size_t)b * SEQ * DM + h * DH;
  const bf16_t* qv_bh = qv + (size_t)b * SEQ * DM + h * DH;
  const bf16_t* k_bh  = kb + (size_t)b * SEQ * DM + h * DH;
  const bf16_t* p_h   = pb + h * DH;
  const bf16_t* v_bh  = vT + (size_t)((b * NH + h) * DH) * SEQ;

  int lm = lane & 15;
  int hi = (lane & 16) ? 1 : 0;
  int half8 = hi ? 8 : 0;

  v16bf quA[2], qvA0[2], qvA1[2];
  {
    int rA  = i0 + lm;
    int rA1 = rA + 1; if (rA1 > SEQ - 1) rA1 = SEQ - 1;
#pragma unroll
    for (int c = 0; c < 2; ++c) {
      quA[c]  = load_afrag(qu_bh, rA,  c, DM, lane);
      qvA0[c] = load_afrag(qv_bh, rA,  c, DM, lane);
      qvA1[c] = load_afrag(qv_bh, rA1, c, DM, lane);  // rows i0+1.. for upper part
    }
  }

  v8f cacc[4];
#pragma unroll
  for (int dt = 0; dt < 4; ++dt) cacc[dt] = (v8f){};
  float m8[8], l8[8];
#pragma unroll
  for (int r = 0; r < 8; ++r) { m8[r] = -3.4e38f; l8[r] = 0.f; }

  float (&Tlo)[16][33] = ldsT[wsl][0];
  float (&Tup)[16][33] = ldsT[wsl][1];
  __bf16 (&Pst)[16][32] = ldsP[wsl];

  for (int j0 = 0; j0 < SEQ; j0 += 32) {
    if (j0 + 32 < SEQ)
      __builtin_prefetch(k_bh + (size_t)(j0 + 32 + lm) * DM, 0, 3);

    v8f scf[2];
#pragma unroll
    for (int sub = 0; sub < 2; ++sub) {
      int jt = j0 + sub * 16;
      // ---- content score: (q+u) . k^T  (K = dh = 64 -> 2 WMMAs) ----
      v8f sc = (v8f){};
      {
        const bf16_t* kr = k_bh + (size_t)(jt + lm) * DM + (hi ? 16 : 0);
        sc = wmma_bf16(quA[0], load_bfrag(kr), sc);
        sc = wmma_bf16(quA[1], load_bfrag(kr + 32), sc);
      }
      // ---- pos score via analytic rel_shift:
      //   m = S-1-i+j ; m==S -> 0 ; m<S -> PS[i,m] ; m>S -> PS[i+1, m-S-1]
      //   in-tile column: cc = 15 - di + dj  (same for both regions)
      bool lowN = (jt <= i0);
      bool upN  = (jt >= i0);
      int mbase = SEQ - 16 - i0 + jt;   // p row for Tlo at cc=0
      int ubase = jt - i0 - 17;         // p row for Tup at cc=0
      if (lowN) {
#pragma unroll
        for (int th = 0; th < 2; ++th) {
          int cc = th * 16 + lm;
          int prow = mbase + cc;
          prow = prow < 0 ? 0 : (prow > SEQ - 1 ? SEQ - 1 : prow);
          const bf16_t* pr = p_h + (size_t)prow * DM + (hi ? 16 : 0);
          v8f T = (v8f){};
          T = wmma_bf16(qvA0[0], load_bfrag(pr), T);
          T = wmma_bf16(qvA0[1], load_bfrag(pr + 32), T);
#pragma unroll
          for (int r = 0; r < 8; ++r) Tlo[r + half8][th * 16 + lm] = T[r];
        }
      }
      if (upN) {
#pragma unroll
        for (int th = 0; th < 2; ++th) {
          int cc = th * 16 + lm;
          int prow = ubase + cc;
          prow = prow < 0 ? 0 : (prow > SEQ - 1 ? SEQ - 1 : prow);
          const bf16_t* pr = p_h + (size_t)prow * DM + (hi ? 16 : 0);
          v8f T = (v8f){};
          T = wmma_bf16(qvA1[0], load_bfrag(pr), T);
          T = wmma_bf16(qvA1[1], load_bfrag(pr + 32), T);
#pragma unroll
          for (int r = 0; r < 8; ++r) Tup[r + half8][th * 16 + lm] = T[r];
        }
      }
      __builtin_amdgcn_wave_barrier();  // order LDS stores before gathers
      // ---- combine + scale 1/sqrt(D) ----
#pragma unroll
      for (int r = 0; r < 8; ++r) {
        int di = r + half8;
        int i  = i0 + di;
        int j  = jt + lm;
        int m  = (SEQ - 1) - i + j;
        int cc = 15 - di + lm;
        float posv = (m == SEQ) ? 0.f : (m < SEQ ? Tlo[di][cc] : Tup[di][cc]);
        sc[r] = (sc[r] + posv) * 0.0625f;  // 1/sqrt(256)
      }
      scf[sub] = sc;
      __builtin_amdgcn_wave_barrier();
    }

    // ---- online softmax over the 32-column chunk ----
    float alpha[8];
#pragma unroll
    for (int r = 0; r < 8; ++r) {
      float mx = fmaxf(scf[0][r], scf[1][r]);
#pragma unroll
      for (int msk = 1; msk < 16; msk <<= 1) mx = fmaxf(mx, __shfl_xor(mx, msk, 32));
      float mnew = fmaxf(m8[r], mx);
      alpha[r] = __expf(m8[r] - mnew);
      float p0 = __expf(scf[0][r] - mnew);
      float p1 = __expf(scf[1][r] - mnew);
      scf[0][r] = p0; scf[1][r] = p1;
      float rs = p0 + p1;
#pragma unroll
      for (int msk = 1; msk < 16; msk <<= 1) rs += __shfl_xor(rs, msk, 32);
      l8[r] = l8[r] * alpha[r] + rs;
      m8[r] = mnew;
    }
#pragma unroll
    for (int dt = 0; dt < 4; ++dt)
#pragma unroll
      for (int r = 0; r < 8; ++r) cacc[dt][r] *= alpha[r];

    // ---- probs: C-frag -> A-frag relayout through LDS ----
#pragma unroll
    for (int r = 0; r < 8; ++r) {
      int di = r + half8;
      Pst[di][lm]      = (bf16_t)scf[0][r];
      Pst[di][16 + lm] = (bf16_t)scf[1][r];
    }
    __builtin_amdgcn_wave_barrier();
    v16bf aP;
    {
      int t0 = hi ? 16 : 0;
      const __bf16* pr = &Pst[lm][0];
      union { uint4 u; __bf16 h[8]; } a0, a1;
      a0.u = *(const uint4*)(pr + t0);
      a1.u = *(const uint4*)(pr + t0 + 8);
#pragma unroll
      for (int e = 0; e < 8; ++e) { aP[e] = a0.h[e]; aP[e + 8] = a1.h[e]; }
    }
    __builtin_amdgcn_wave_barrier();
    // ---- ctx += P @ V   (K = 32 = this j-chunk; 4 x 16-wide d tiles) ----
#pragma unroll
    for (int dt = 0; dt < 4; ++dt) {
      const bf16_t* vr = v_bh + (size_t)(dt * 16 + lm) * SEQ + j0 + (hi ? 16 : 0);
      cacc[dt] = wmma_bf16(aP, load_bfrag(vr), cacc[dt]);
    }
  }

  // ---- finalize: divide by row sums, store ctx bf16 [B*S, 256] ----
#pragma unroll
  for (int r = 0; r < 8; ++r) {
    float inv = 1.f / l8[r];
    int row = b * SEQ + i0 + r + half8;
#pragma unroll
    for (int dt = 0; dt < 4; ++dt) {
      int col = h * DH + dt * 16 + lm;
      ctx[(size_t)row * DM + col] = (bf16_t)(cacc[dt][r] * inv);
    }
  }
}

// ---------------------------------------------------------------------------
extern "C" void kernel_launch(void* const* d_in, const int* in_sizes, int n_in,
                              void* d_out, int out_size, void* d_ws, size_t ws_size,
                              hipStream_t stream) {
  const float* x     = (const float*)d_in[0];
  const float* gamma = (const float*)d_in[1];
  const float* beta  = (const float*)d_in[2];
  const float* Wq    = (const float*)d_in[3];
  const float* bq    = (const float*)d_in[4];
  const float* Wk    = (const float*)d_in[5];
  const float* bk    = (const float*)d_in[6];
  const float* Wv    = (const float*)d_in[7];
  const float* bv    = (const float*)d_in[8];
  const float* Wpos  = (const float*)d_in[9];
  const float* ub    = (const float*)d_in[10];  // [4,64] flat == per-column D bias
  const float* vb    = (const float*)d_in[11];
  const float* Wout  = (const float*)d_in[12];
  const float* bout  = (const float*)d_in[13];
  float* out = (float*)d_out;

  char* ws = (char*)d_ws;
  size_t o = 0;
  const size_t ACT = (size_t)ROWS * DM * sizeof(bf16_t);   // 8 MB
  const size_t PEB = (size_t)SEQ * DM * sizeof(bf16_t);    // 1 MB
  const size_t WTB = (size_t)DM * DM * sizeof(bf16_t);     // 128 KB
  bf16_t* xln   = (bf16_t*)(ws + o); o += ACT;
  bf16_t* peb   = (bf16_t*)(ws + o); o += PEB;
  bf16_t* wqT   = (bf16_t*)(ws + o); o += WTB;
  bf16_t* wkT   = (bf16_t*)(ws + o); o += WTB;
  bf16_t* wvT   = (bf16_t*)(ws + o); o += WTB;
  bf16_t* wposT = (bf16_t*)(ws + o); o += WTB;
  bf16_t* woutT = (bf16_t*)(ws + o); o += WTB;
  bf16_t* qub   = (bf16_t*)(ws + o); o += ACT;
  bf16_t* qvb   = (bf16_t*)(ws + o); o += ACT;
  bf16_t* kbuf  = (bf16_t*)(ws + o); o += ACT;
  bf16_t* vTb   = (bf16_t*)(ws + o); o += ACT;
  bf16_t* pbuf  = (bf16_t*)(ws + o); o += PEB;
  bf16_t* ctxb  = (bf16_t*)(ws + o); o += ACT;   // total ~53 MB

  ln_kernel<<<2048, 256, 0, stream>>>(x, gamma, beta, xln);
  pe_kernel<<<2048, 256, 0, stream>>>(peb);
  dim3 gw(256, 5);
  wtrans_kernel<<<gw, 256, 0, stream>>>(Wq, Wk, Wv, Wpos, Wout,
                                        wqT, wkT, wvT, wposT, woutT);
  // q (+u and +v biased copies), k, v (transposed), pos projection
  proj_gemm_kernel<<<4096, 128, 0, stream>>>(xln, wqT, bq, ub, vb,
                                             qub, qvb, nullptr, ROWS, 0);
  proj_gemm_kernel<<<4096, 128, 0, stream>>>(xln, wkT, bk, nullptr, nullptr,
                                             kbuf, nullptr, nullptr, ROWS, 0);
  proj_gemm_kernel<<<4096, 128, 0, stream>>>(xln, wvT, bv, nullptr, nullptr,
                                             vTb, nullptr, nullptr, ROWS, 1);
  proj_gemm_kernel<<<512, 128, 0, stream>>>(peb, wposT, nullptr, nullptr, nullptr,
                                            pbuf, nullptr, nullptr, SEQ, 0);
  // fused flash attention with analytic rel_shift
  attn_kernel<<<1024, 128, 0, stream>>>(qub, qvb, kbuf, vTb, pbuf, ctxb);
  // output projection -> fp32
  proj_gemm_kernel<<<4096, 128, 0, stream>>>(ctxb, woutT, bout, nullptr, nullptr,
                                             nullptr, nullptr, out, ROWS, 0);
}
